// Cvx_KnapsackNet_29600914604059
// MI455X (gfx1250) — matrix operand order repaired and checked by reading, not compiled, not run on hardware
//
#include <hip/hip_runtime.h>

// ---------------- types / WMMA wrappers (CDNA5 gfx1250, wave32) ----------------
typedef float    v8f   __attribute__((ext_vector_type(8)));
typedef float    v2f   __attribute__((ext_vector_type(2)));
typedef __bf16   v16bf __attribute__((ext_vector_type(16)));
typedef unsigned v4u   __attribute__((ext_vector_type(4)));
typedef int      v8i   __attribute__((ext_vector_type(8)));
typedef int      v4i   __attribute__((ext_vector_type(4)));

__device__ __forceinline__ v8f wmma_f32(v2f a, v2f b, v8f c) {
  // V_WMMA_F32_16X16X4_F32 : exact-precision f32 matrix op
  return __builtin_amdgcn_wmma_f32_16x16x4_f32(false, a, false, b, (short)0, c, false, false);
}
__device__ __forceinline__ v8f wmma_bf16(v16bf a, v16bf b, v8f c) {
  // V_WMMA_F32_16X16X32_BF16 : bf16 in, f32 accumulate
  return __builtin_amdgcn_wmma_f32_16x16x32_bf16(false, a, false, b, (short)0, c, false, false);
}

// ---------------- problem constants ----------------
constexpr int NC = 32;     // context
constexpr int NH = 3200;   // hidden
constexpr int NR = 500;    // resources
constexpr int NK = 30;     // constraints
constexpr int ITERS = 200;

// workspace layout (floats)
constexpr size_t OFF_WP   = 0;                         // Wp   [32][512] padded W
constexpr size_t OFF_WT   = OFF_WP   + 32ull*512;      // WT   [512][32] = Wp^T
constexpr size_t OFF_SINV = OFF_WT   + 512ull*32;      // Sinv [32][32]
constexpr size_t OFF_WA   = OFF_SINV + 32ull*32;       // wa = cost padded [1024][512]
constexpr size_t OFF_ZA   = OFF_WA   + 1024ull*512;
constexpr size_t OFF_UA   = OFF_ZA   + 1024ull*512;
constexpr size_t OFF_ZC   = OFF_UA   + 1024ull*512;
constexpr size_t OFF_UC   = OFF_ZC   + 1024ull*512;
constexpr size_t OFF_ZB   = OFF_UC   + 1024ull*512;    // [1024][32]
constexpr size_t OFF_UB   = OFF_ZB   + 1024ull*32;
constexpr size_t OFF_BF   = OFF_UB   + 1024ull*32;     // bf16 region below (reinterpret)

// =====================================================================
// Tensor Data Mover: 2D tile load (global bf16, row-major, stride 3200 elem)
// tile = 32 (K) x 128 (rows); LDS pad 1 DWORD per 16 DWORDs -> row stride 34 bf16
// =====================================================================
__device__ __forceinline__ unsigned lds_off32(const void* p) {
  // low 32 bits of a generic pointer into LDS == LDS byte offset (ISA 10.2)
  return (unsigned)(unsigned long long)(uintptr_t)p;
}

__device__ __forceinline__ void tdm_load_tile(unsigned ldsaddr, const __bf16* gptr,
                                              unsigned tensor_rows) {
  unsigned long long ga = (unsigned long long)(uintptr_t)gptr;
  v4u g0;
  g0.x = 1u;                                            // count=1, is_restore=0, gather=0
  g0.y = ldsaddr;                                       // lds_addr
  g0.z = (unsigned)(ga & 0xffffffffu);                  // global_addr[31:0]
  g0.w = (unsigned)((ga >> 32) & 0x1ffffffu) | (2u << 30); // global_addr[56:32], type=2
  v8i g1;
  // data_size=1(2B)<<16 | pad_enable<<20 | pad_interval=3(16 DW)<<22 | pad_amount=0(1 DW)<<25
  g1[0] = (int)((1u << 16) | (1u << 20) | (3u << 22));
  unsigned td0 = (unsigned)NH;                          // tensor_dim0 (K length)
  g1[1] = (int)((td0 & 0xffffu) << 16);                 // [63:48] = tensor_dim0[15:0]
  g1[2] = (int)(((td0 >> 16) & 0xffffu) | ((tensor_rows & 0xffffu) << 16));
  g1[3] = (int)(((tensor_rows >> 16) & 0xffffu) | (32u << 16)); // tile_dim0 = 32
  g1[4] = (int)(128u);                                  // tile_dim1 = 128, tile_dim2 = 0
  g1[5] = (int)((unsigned)NH);                          // tensor_dim0_stride[31:0]
  g1[6] = 0;
  g1[7] = 0;
  v4i gz4 = {0, 0, 0, 0};
  v8i gz8 = {0, 0, 0, 0, 0, 0, 0, 0};
  // 6-arg form (clang-23 / therock-10.0 HIP headers)
  __builtin_amdgcn_tensor_load_to_lds(g0, g1, gz4, gz4, gz8, 0);
}

// =====================================================================
// Prep: pad W -> Wp[32][512], WT[512][32]
// =====================================================================
__global__ void __launch_bounds__(256)
k_pad(const float* __restrict__ W, float* __restrict__ Wp, float* __restrict__ WT) {
  int idx = blockIdx.x * 256 + threadIdx.x;
  if (idx >= 32 * 512) return;
  int j = idx >> 9, k = idx & 511;
  float v = (j < NK && k < NR) ? W[j * NR + k] : 0.f;
  Wp[j * 512 + k] = v;
  WT[k * 32 + j]  = v;
}

// =====================================================================
// Prep: S = I + 0.5 W W^T (32x32 padded, SPD), invert via Gauss-Jordan
// =====================================================================
__global__ void __launch_bounds__(256)
k_sinv(const float* __restrict__ Wp, float* __restrict__ Sinv) {
  __shared__ float sM[32][65];
  __shared__ float sFac[32];
  int tid = threadIdx.x;
  for (int e = tid; e < 32 * 32; e += 256) {
    int i = e >> 5, j = e & 31;
    float s = 0.f;
    for (int k = 0; k < 512; ++k) s += Wp[i * 512 + k] * Wp[j * 512 + k];
    sM[i][j]      = 0.5f * s + (i == j ? 1.f : 0.f);
    sM[i][32 + j] = (i == j) ? 1.f : 0.f;
  }
  __syncthreads();
  for (int p = 0; p < 32; ++p) {
    float pv = sM[p][p];
    __syncthreads();
    if (tid < 64) sM[p][tid] = sM[p][tid] / pv;
    __syncthreads();
    if (tid < 32) sFac[tid] = (tid == p) ? 0.f : sM[tid][p];
    __syncthreads();
    for (int e = tid; e < 32 * 64; e += 256) {
      int r = e >> 6, c = e & 63;
      if (r != p) sM[r][c] -= sFac[r] * sM[p][c];
    }
    __syncthreads();
  }
  for (int e = tid; e < 32 * 32; e += 256) {
    int i = e >> 5, j = e & 31;
    Sinv[i * 32 + j] = sM[i][32 + j];
  }
}

// =====================================================================
// Prep: one-time f32 -> bf16 weight conversions (amortized across all tiles)
// =====================================================================
__global__ void __launch_bounds__(256)
k_cvt(const float* __restrict__ src, __bf16* __restrict__ dst, int n) {
  int i = blockIdx.x * 256 + threadIdx.x;
  if (i < n) dst[i] = (__bf16)src[i];
}
__global__ void __launch_bounds__(256)
k_cvt_pad512(const float* __restrict__ src, __bf16* __restrict__ dst) {
  int i = blockIdx.x * 256 + threadIdx.x;            // [512][3200], rows >= 500 zero
  if (i >= 512 * NH) return;
  int r = i / NH;
  dst[i] = (r < NR) ? (__bf16)src[i] : (__bf16)0.f;
}

// =====================================================================
// Layer 1: h1 = lrelu(d @ W1^T + b1)  M=1024 N=3200 K=32, bf16 output
// =====================================================================
__global__ void __launch_bounds__(256)
k_layer1(const float* __restrict__ d, const float* __restrict__ W1,
         const float* __restrict__ b1, __bf16* __restrict__ h1h) {
  int wave = threadIdx.x >> 5, lane = threadIdx.x & 31;
  int tile = blockIdx.x * 8 + wave;       // 64 x 200 = 12800 tiles
  int tm = tile & 63, tn = tile >> 6;
  int m0 = tm * 16, n0 = tn * 16;
  int row = lane & 15, hl = lane >> 4;

  v16bf a, b;
  const float* drow = d + (size_t)(m0 + row) * NC;
  int kbA = hl * 16;
#pragma unroll
  for (int i = 0; i < 8; ++i) {           // A 16x32: lanes 0-15 K0-15; 16-31 K16-31
    int kk = kbA + 2 * i;
    a[2 * i]     = (__bf16)drow[kk];
    a[2 * i + 1] = (__bf16)drow[kk + 1];
  }
  const float* wrow = W1 + (size_t)(n0 + row) * NC;   // B[k][n] = W1[n][k]
  int kbB = hl * 16;
#pragma unroll
  for (int i = 0; i < 8; ++i) {
    b[2 * i]     = (__bf16)wrow[kbB + 2 * i];
    b[2 * i + 1] = (__bf16)wrow[kbB + 2 * i + 1];
  }
  v8f c = {};
  c = wmma_bf16(a, b, c);
  float bias = b1[n0 + row];
#pragma unroll
  for (int j = 0; j < 8; ++j) {           // C: vgpr j -> M = j (+8 upper lanes), N = lane&15
    int m = m0 + j + hl * 8;
    float v = c[j] + bias;
    v = v > 0.f ? v : 0.1f * v;
    h1h[(size_t)m * NH + n0 + row] = (__bf16)v;
  }
}

// =====================================================================
// Layers 2/3: out = act(A[1024,3200]bf16 @ B[Brows,3200]bf16^T + bias)
// 128x128 tile, 8 waves (2x4), TDM double-buffered LDS staging.
// LDS tiles row-major [row][34] (TDM pad); WMMA fragment gather transposes B.
// =====================================================================
__global__ void __launch_bounds__(256)
k_mlp(const __bf16* __restrict__ A, const __bf16* __restrict__ Bm,
      const float* __restrict__ bias, float* __restrict__ outF,
      __bf16* __restrict__ outH, int Brows, int Nreal, int ostride, int do_lrelu) {
  __shared__ __bf16 sA[2][128][34];
  __shared__ __bf16 sB[2][128][34];
  int tid = threadIdx.x;
  int wave = tid >> 5, lane = tid & 31;
  int tm = blockIdx.x & 7, tn = blockIdx.x >> 3;
  int m0 = tm * 128, n0 = tn * 128;
  int wr = wave >> 2, wc = wave & 3;
  int row = lane & 15, hl = lane >> 4;
  constexpr int NCH = NH / 32;            // 100 K-chunks

  const __bf16* Abase = A + (size_t)m0 * NH;
  const __bf16* Bbase = Bm + (size_t)n0 * NH;

  if (wave == 0) {                        // prologue: DMA chunk 0 into buffer 0
    tdm_load_tile(lds_off32(&sA[0][0][0]), Abase, 1024u);
    tdm_load_tile(lds_off32(&sB[0][0][0]), Bbase, (unsigned)Brows);
  }

  v8f acc[4][2] = {};
  for (int c = 0; c < NCH; ++c) {
    int buf = c & 1;
    if (wave == 0) {
      if (c + 1 < NCH) {                  // prefetch next chunk into other buffer
        int k1 = (c + 1) * 32;
        tdm_load_tile(lds_off32(&sA[buf ^ 1][0][0]), Abase + k1, 1024u);
        tdm_load_tile(lds_off32(&sB[buf ^ 1][0][0]), Bbase + k1, (unsigned)Brows);
        __builtin_amdgcn_s_wait_tensorcnt(2);   // current chunk's pair complete
      } else {
        __builtin_amdgcn_s_wait_tensorcnt(0);
      }
    }
    __syncthreads();                      // LDS tiles ready for all waves

    v16bf afr[4], bfr[2];
    int kbA = hl * 16, kbB = hl * 16;
#pragma unroll
    for (int sm = 0; sm < 4; ++sm) {
      int mr = wr * 64 + sm * 16 + row;
#pragma unroll
      for (int i = 0; i < 8; ++i) {
        int kk = kbA + 2 * i;
        afr[sm][2 * i]     = sA[buf][mr][kk];
        afr[sm][2 * i + 1] = sA[buf][mr][kk + 1];
      }
    }
#pragma unroll
    for (int sn = 0; sn < 2; ++sn) {
      int nc = wc * 32 + sn * 16 + row;   // B[k][n] = Bm[n][k]: contiguous k per lane
#pragma unroll
      for (int i = 0; i < 8; ++i) {
        bfr[sn][2 * i]     = sB[buf][nc][kbB + 2 * i];
        bfr[sn][2 * i + 1] = sB[buf][nc][kbB + 2 * i + 1];
      }
    }
#pragma unroll
    for (int sm = 0; sm < 4; ++sm)
#pragma unroll
      for (int sn = 0; sn < 2; ++sn)
        acc[sm][sn] = wmma_bf16(afr[sm], bfr[sn], acc[sm][sn]);
    __syncthreads();                      // buffer reusable at c+2
  }

  // epilogue
#pragma unroll
  for (int sm = 0; sm < 4; ++sm)
    for (int sn = 0; sn < 2; ++sn) {
      int gn = n0 + wc * 32 + sn * 16 + row;
      float bv = (gn < Nreal) ? bias[gn] : 0.f;
#pragma unroll
      for (int j = 0; j < 8; ++j) {
        int gm = m0 + wr * 64 + sm * 16 + j + hl * 8;
        float v = acc[sm][sn][j] + bv;
        if (do_lrelu) v = v > 0.f ? v : 0.1f * v;
        if (gn >= Nreal) v = 0.f;         // zero padding (cost buffer)
        if (gn < ostride) {
          if (outH) outH[(size_t)gm * ostride + gn] = (__bf16)v;
          else      outF[(size_t)gm * ostride + gn] = v;
        }
      }
    }
}

// =====================================================================
// ADMM: persistent per-block solver, 16 batch rows/block, 200 iterations,
// Schur-complement solve (I + 0.5 W W^T)^{-1}; f32 WMMA 16x16x4 thin GEMMs.
// =====================================================================
__global__ void __launch_bounds__(128)
k_admm(const float* __restrict__ cap, const float* __restrict__ WpG,
       const float* __restrict__ WTG, const float* __restrict__ SinvG,
       const float* __restrict__ wa,
       float* za, float* ua, float* zc, float* uc, float* zb, float* ub,
       float* __restrict__ out) {
  __shared__ float s_q[16][516];
  __shared__ float s_t4[4][16][32];
  __shared__ float s_y1[16][32];
  __shared__ float s_Sinv[32][33];
  int tid = threadIdx.x, wave = tid >> 5, lane = tid & 31;
  int r0 = blockIdx.x * 16;
  int row = lane & 15, hl = lane >> 4;
  int kb2 = hl * 2;                       // f32 16x16x4: lanes 0-15 K0/1, 16-31 K2/3

  for (int e = tid; e < 16 * 512; e += 128) {
    int g = (r0 + (e >> 9)) * 512 + (e & 511);
    za[g] = 0.f; ua[g] = 0.f; zc[g] = 0.f; uc[g] = 0.f;
  }
  for (int e = tid; e < 16 * 32; e += 128) {
    int g = (r0 + (e >> 5)) * 32 + (e & 31);
    zb[g] = 0.f; ub[g] = 0.f;
  }
  for (int e = tid; e < 32 * 32; e += 128) s_Sinv[e >> 5][e & 31] = SinvG[e];
  __threadfence();
  __syncthreads();

  for (int it = 0; it < ITERS; ++it) {
    // q = 0.5*(va - vc) + 1 (masked)
    for (int e = tid; e < 16 * 512; e += 128) {
      int r = e >> 9, c = e & 511;
      int g = (r0 + r) * 512 + c;
      float q = 0.f;
      if (c < NR) {
        float va = wa[g] + za[g] - ua[g];
        float vc = zc[g] - uc[g];
        q = 0.5f * (va - vc) + 1.0f;
      }
      s_q[r][c] = q;
    }
    __syncthreads();

    // GEMM1: t_partial = q @ W^T ; K split over 4 waves
    v8f t0 = {}, t1 = {};
    for (int k0 = wave * 128; k0 < wave * 128 + 128; k0 += 4) {
      v2f a; a.x = s_q[row][k0 + kb2]; a.y = s_q[row][k0 + kb2 + 1];
      v2f b0, b1;
      b0.x = WTG[(k0 + kb2) * 32 + row];       b0.y = WTG[(k0 + kb2 + 1) * 32 + row];
      b1.x = WTG[(k0 + kb2) * 32 + 16 + row];  b1.y = WTG[(k0 + kb2 + 1) * 32 + 16 + row];
      t0 = wmma_f32(a, b0, t0);
      t1 = wmma_f32(a, b1, t1);
    }
#pragma unroll
    for (int j = 0; j < 8; ++j) {
      int m = j + hl * 8;
      s_t4[wave][m][row]      = t0[j];
      s_t4[wave][m][16 + row] = t1[j];
    }
    __syncthreads();

    // reduce partials, add vb - 2*cap
    for (int e = tid; e < 16 * 32; e += 128) {
      int m = e >> 5, j = e & 31;
      float t = s_t4[0][m][j] + s_t4[1][m][j] + s_t4[2][m][j] + s_t4[3][m][j];
      if (j < NK) {
        int g = (r0 + m) * 32 + j;
        t += (zb[g] - ub[g]) - 2.0f * cap[j];
      } else t = 0.f;
      s_t4[0][m][j] = t;
    }
    __syncthreads();

    // GEMM2 (wave 0): y1 = t @ Sinv
    if (wave == 0) {
      v8f y0 = {}, y1 = {};
      for (int k0 = 0; k0 < 32; k0 += 4) {
        v2f a; a.x = s_t4[0][row][k0 + kb2]; a.y = s_t4[0][row][k0 + kb2 + 1];
        v2f b0, b1;
        b0.x = s_Sinv[k0 + kb2][row];       b0.y = s_Sinv[k0 + kb2 + 1][row];
        b1.x = s_Sinv[k0 + kb2][16 + row];  b1.y = s_Sinv[k0 + kb2 + 1][16 + row];
        y0 = wmma_f32(a, b0, y0);
        y1 = wmma_f32(a, b1, y1);
      }
#pragma unroll
      for (int j = 0; j < 8; ++j) {
        int m = j + hl * 8;
        s_y1[m][row]      = y0[j];
        s_y1[m][16 + row] = y1[j];
      }
    }
    __syncthreads();

    // b-part update
    for (int e = tid; e < 16 * 32; e += 128) {
      int m = e >> 5, j = e & 31;
      int g = (r0 + m) * 32 + j;
      float vb = (j < NK) ? (zb[g] - ub[g]) : 0.f;
      float xb = 0.5f * (vb - s_y1[m][j]);
      float uv = ub[g];
      float nz = fmaxf(xb + uv, 0.f);
      zb[g] = nz; ub[g] = uv + xb - nz;
      if (it == ITERS - 1 && j < NK) out[(size_t)(r0 + m) * 1030 + 500 + j] = xb;
    }

    // GEMM3 + fused epilogue: p = y1 @ Wp ; 32 N-tiles over 4 waves
    for (int i = 0; i < 8; ++i) {
      int n0 = (wave + 4 * i) * 16;
      v8f p = {};
#pragma unroll
      for (int k0 = 0; k0 < 32; k0 += 4) {
        v2f a; a.x = s_y1[row][k0 + kb2]; a.y = s_y1[row][k0 + kb2 + 1];
        v2f b; b.x = WpG[(k0 + kb2) * 512 + n0 + row];
               b.y = WpG[(k0 + kb2 + 1) * 512 + n0 + row];
        p = wmma_f32(a, b, p);
      }
      int gn = n0 + row;
      bool mk = gn < NR;
#pragma unroll
      for (int j = 0; j < 8; ++j) {
        int m = j + hl * 8;
        int g = (r0 + m) * 512 + gn;
        float zav = za[g], uav = ua[g], zcv = zc[g], ucv = uc[g];
        float va = mk ? (wa[g] + zav - uav) : 0.f;
        float vc = mk ? (zcv - ucv) : 0.f;
        float r2 = mk ? (va + vc - 2.0f) : 0.f;
        float pv = mk ? p[j] : 0.f;
        float y2 = 0.5f * (r2 - pv);
        float xa = 0.5f * (va - pv - y2);
        float xc = 0.5f * (vc - y2);
        float nza = fmaxf(xa + uav, 0.f);
        float nzc = fmaxf(xc + ucv, 0.f);
        za[g] = nza; ua[g] = uav + xa - nza;
        zc[g] = nzc; uc[g] = ucv + xc - nzc;
        if (it == ITERS - 1 && mk) {
          out[(size_t)(r0 + m) * 1030 + gn]       = xa;
          out[(size_t)(r0 + m) * 1030 + 530 + gn] = xc;
        }
      }
    }
    __threadfence();
    __syncthreads();
  }
}

// =====================================================================
extern "C" void kernel_launch(void* const* d_in, const int* in_sizes, int n_in,
                              void* d_out, int out_size, void* d_ws, size_t ws_size,
                              hipStream_t stream) {
  (void)in_sizes; (void)n_in; (void)out_size; (void)ws_size;
  const float* d   = (const float*)d_in[0];
  const float* W1  = (const float*)d_in[1];
  const float* b1  = (const float*)d_in[2];
  const float* W2  = (const float*)d_in[3];
  const float* b2  = (const float*)d_in[4];
  const float* W3  = (const float*)d_in[5];
  const float* b3  = (const float*)d_in[6];
  const float* Wm  = (const float*)d_in[7];
  const float* cap = (const float*)d_in[8];

  float* ws   = (float*)d_ws;
  float* Wp   = ws + OFF_WP;
  float* WT   = ws + OFF_WT;
  float* Sinv = ws + OFF_SINV;
  float* wa   = ws + OFF_WA;
  float* za   = ws + OFF_ZA;
  float* ua   = ws + OFF_UA;
  float* zc   = ws + OFF_ZC;
  float* uc   = ws + OFF_UC;
  float* zb   = ws + OFF_ZB;
  float* ub   = ws + OFF_UB;

  __bf16* bfb = (__bf16*)(ws + OFF_BF);
  __bf16* h1h = bfb;                                  // [1024][3200]
  __bf16* h2h = h1h + 1024ull * NH;                   // [1024][3200]
  __bf16* W2h = h2h + 1024ull * NH;                   // [3200][3200]
  __bf16* W3h = W2h + (size_t)NH * NH;                // [512][3200] padded

  k_pad      <<<64,    256, 0, stream>>>(Wm, Wp, WT);
  k_sinv     <<<1,     256, 0, stream>>>(Wp, Sinv);
  k_cvt      <<<40000, 256, 0, stream>>>(W2, W2h, NH * NH);
  k_cvt_pad512<<<6400, 256, 0, stream>>>(W3, W3h);
  k_layer1   <<<1600,  256, 0, stream>>>(d, W1, b1, h1h);
  k_mlp      <<<200,   256, 0, stream>>>(h1h, W2h, b2, nullptr, h2h, NH, NH, NH, 1);
  k_mlp      <<<32,    256, 0, stream>>>(h2h, W3h, b3, wa, nullptr, 512, NR, 512, 0);
  k_admm     <<<64,    128, 0, stream>>>(cap, Wp, WT, Sinv, wa,
                                         za, ua, zc, uc, zb, ub, (float*)d_out);
}
